// DecoderOnlyTransformer_76768245449400
// MI455X (gfx1250) — matrix operand order repaired and checked by reading, not compiled, not run on hardware
//
#include <hip/hip_runtime.h>

typedef __bf16 bf16;
typedef __attribute__((ext_vector_type(8)))  bf16  v8bf;
typedef __attribute__((ext_vector_type(16))) bf16  v16bf;
typedef __attribute__((ext_vector_type(8)))  float v8f;

#define DEV static __device__ __forceinline__

DEV v8f vzero8() {
    v8f z;
#pragma unroll
    for (int i = 0; i < 8; ++i) z[i] = 0.0f;
    return z;
}

// D = A(16x32) * B(32x16) + C  via V_WMMA_F32_16X16X32_BF16
DEV v8f wmma_bf16(v16bf a, v16bf b, v8f c) {
    return __builtin_amdgcn_wmma_f32_16x16x32_bf16(
        /*neg_a=*/false, a, /*neg_b=*/false, b,
        /*c_mod=*/(short)0, c, /*reuse_a=*/false, /*reuse_b=*/false);
}

// A-matrix fragment (16x32 bf16, row-major source, stride lda elements).
// Layout used here: lane<16 -> row M=lane,    K = 0..15 (contiguous)
//             lane>=16 -> row M=lane-16, K = 16..31
DEV v16bf load_a_frag(const bf16* __restrict__ A, int lda, int m0, int k0, int lane) {
    int row = m0 + (lane & 15);
    int kh  = lane >> 4;
    const bf16* p = A + (size_t)row * lda + k0 + kh * 16;
    return *reinterpret_cast<const v16bf*>(p);
}

// B-matrix fragment (32x16). B[k][n] = W[n][k], W row-major [N,K] stride ldw.
// Layout used here: lane holds column n = lane&15; K-range = (lane>>4)*16 .. +15 (contiguous).
DEV v16bf load_b_frag(const bf16* __restrict__ W, int ldw, int n0, int k0, int lane, int nmax) {
    int n = n0 + (lane & 15);
    if (n >= nmax) n = nmax - 1;           // clamp for ragged N (logits)
    int kh = lane >> 4;
    const bf16* p = W + (size_t)n * ldw + k0 + kh * 16;
    return *reinterpret_cast<const v16bf*>(p);
}

// ---------------------------------------------------------------------------
// fp32 -> bf16 conversion
__global__ __launch_bounds__(256) void cvt_kernel(const float* __restrict__ in,
                                                  bf16* __restrict__ out, size_t n) {
    size_t i = (size_t)blockIdx.x * 256 + threadIdx.x;
    if (i < n) out[i] = (bf16)in[i];
}

// x[b,s,:] = wte[idx[b,s],:] + wpe[s,:]
__global__ __launch_bounds__(256) void embed_kernel(const int* __restrict__ idx,
                                                    const float* __restrict__ wte,
                                                    const float* __restrict__ wpe,
                                                    float* __restrict__ x, int S, int D) {
    int i = blockIdx.x * 256 + threadIdx.x;   // over B*S*D
    int d = i % D;
    int t = i / D;
    int s = t % S;
    x[i] = wte[(size_t)idx[t] * D + d] + wpe[(size_t)s * D + d];
}

// LayerNorm: one wave per token, fp32 in (kept), bf16 out.
__global__ __launch_bounds__(256) void ln_kernel(const float* __restrict__ x,
                                                 const float* __restrict__ w,
                                                 const float* __restrict__ b,
                                                 bf16* __restrict__ out, int D) {
    int lane = threadIdx.x & 31;
    int wave = threadIdx.x >> 5;
    int tok  = blockIdx.x * 8 + wave;
    const float* xr = x + (size_t)tok * D;

    float s = 0.0f;
    for (int d = lane; d < D; d += 32) s += xr[d];
#pragma unroll
    for (int m = 1; m < 32; m <<= 1) s += __shfl_xor(s, m, 32);
    float mu = s / (float)D;

    float v = 0.0f;
    for (int d = lane; d < D; d += 32) { float t = xr[d] - mu; v += t * t; }
#pragma unroll
    for (int m = 1; m < 32; m <<= 1) v += __shfl_xor(v, m, 32);
    float rstd = rsqrtf(v / (float)D + 1e-5f);

    bf16* orow = out + (size_t)tok * D;
    for (int d = lane; d < D; d += 32)
        orow[d] = (bf16)((xr[d] - mu) * rstd * w[d] + b[d]);
}

// vt[(b*H+h)*64 + d][s] = qkv[b, s, 2D + h*64 + d]   (pre-transpose V for P@V B-frags)
__global__ __launch_bounds__(256) void vtrans_kernel(const bf16* __restrict__ qkv,
                                                     bf16* __restrict__ vt, int S) {
    int i = blockIdx.x * 256 + threadIdx.x;   // over B*H*64*S
    int s    = i % S;
    int rest = i / S;
    int d    = rest % 64;
    int bh   = rest / 64;
    int h = bh % 12, b = bh / 12;
    vt[i] = qkv[((size_t)(b * S + s)) * 2304 + 1536 + h * 64 + d];
}

// ---------------------------------------------------------------------------
// Tiled GEMM: C[M,N] = A[M,K] * W[N,K]^T, bf16 in, f32 accumulate via WMMA.
// Block = 256 threads (8 waves); block tile 256x64; each wave computes a 32x64
// strip (2 M sub-tiles x 4 N tiles = 8 WMMAs per 32-wide K step).
// B tile (64x32 bf16 = 4KB) staged in LDS via async global->LDS DMA, double
// buffered so the next tile's DMA overlaps the current tile's WMMAs.
// epi: 0 = store bf16, 1 = store f32, 2 = residual add into f32, 3 = GELU -> bf16
__global__ __launch_bounds__(256) void gemm_bf16_kernel(const bf16* __restrict__ A, int lda,
                                                        const bf16* __restrict__ W, int ldw,
                                                        int N, int K,
                                                        float* __restrict__ outF,
                                                        bf16* __restrict__ outB,
                                                        int ldo, int epi) {
    __shared__ __align__(32) bf16 Wl[2][64][32];

    int tid  = threadIdx.x;
    int lane = tid & 31;
    int wave = tid >> 5;
    int m0 = blockIdx.y * 256 + wave * 32;
    int n0 = blockIdx.x * 64;

    // Cooperative async B-tile loader: thread t moves 16B; row = t/4, chunk = t%4.
    int lrow   = tid >> 2;       // 0..63
    int lchunk = tid & 3;        // 0..3  (8 bf16 each)
    int wrow = n0 + lrow;
    if (wrow >= N) wrow = N - 1;                       // clamp for ragged N
    const bf16* gWrow = W + (size_t)wrow * ldw + lchunk * 8;
    bf16* ldsDst = &Wl[0][lrow][lchunk * 8];

    auto prefetchB = [&](int buf, int k0) {
        unsigned ldsoff = (unsigned)(unsigned long long)(ldsDst + buf * 64 * 32);
        const bf16* g = gWrow + k0;
        // GLOBAL_LOAD_ASYNC_TO_LDS_B128: vdst = per-lane LDS byte address,
        // vaddr = per-lane 64-bit global address; tracked by ASYNCcnt.
        asm volatile("global_load_async_to_lds_b128 %0, %1, off"
                     :: "v"(ldsoff), "v"(g)
                     : "memory");
    };

    v8f acc[2][4];
#pragma unroll
    for (int u = 0; u < 2; ++u)
#pragma unroll
        for (int t = 0; t < 4; ++t) acc[u][t] = vzero8();

    prefetchB(0, 0);
    for (int k0 = 0, it = 0; k0 < K; k0 += 32, ++it) {
        int cur = it & 1;
        asm volatile("s_wait_asynccnt 0x0" ::: "memory");  // this wave's DMA slice landed
        __syncthreads();                                   // all slices landed; prev reads done
        if (k0 + 32 < K) {
            prefetchB(cur ^ 1, k0 + 32);                   // overlap next DMA with compute
            __builtin_prefetch(A + (size_t)(m0 + (lane & 15)) * lda + k0 + 32, 0, 1);
        }

        v16bf a0 = load_a_frag(A, lda, m0,      k0, lane);
        v16bf a1 = load_a_frag(A, lda, m0 + 16, k0, lane);
#pragma unroll
        for (int t = 0; t < 4; ++t) {
            v16bf bfrag = load_b_frag(&Wl[cur][0][0], 32, 16 * t, 0, lane, 64);
            acc[0][t] = wmma_bf16(a0, bfrag, acc[0][t]);
            acc[1][t] = wmma_bf16(a1, bfrag, acc[1][t]);
        }
    }

    int col = lane & 15, half = lane >> 4;
#pragma unroll
    for (int u = 0; u < 2; ++u) {
#pragma unroll
        for (int t = 0; t < 4; ++t) {
            int n = n0 + 16 * t + col;
            if (n >= N) continue;
#pragma unroll
            for (int r = 0; r < 8; ++r) {
                int m = m0 + u * 16 + r + half * 8;
                float v = acc[u][t][r];
                size_t o = (size_t)m * ldo + n;
                if (epi == 0)      outB[o] = (bf16)v;
                else if (epi == 1) outF[o] = v;
                else if (epi == 2) outF[o] += v;
                else {  // exact GELU
                    float g = 0.5f * v * (1.0f + erff(v * 0.70710678118f));
                    outB[o] = (bf16)g;
                }
            }
        }
    }
}

// ---------------------------------------------------------------------------
// Flash attention: one wave per (b, h, 16-query tile); streams 32-key blocks.
// S = Q(16x64)*K^T via 2x2 WMMA, online softmax, P restaged via LDS, O += P*V.
__global__ __launch_bounds__(128) void attn_kernel(const bf16* __restrict__ qkv,
                                                   const bf16* __restrict__ vt,
                                                   bf16* __restrict__ ctx, int S) {
    __shared__ __align__(32) bf16 Pst[4][16][32];

    const int D3 = 2304, Dm = 768, HD = 64, H = 12;
    int lane = threadIdx.x & 31;
    int wave = threadIdx.x >> 5;
    int q0 = (blockIdx.x * 4 + wave) * 16;
    int h  = blockIdx.y;
    int b  = blockIdx.z;

    const bf16* Q  = qkv + (size_t)b * S * D3 + h * HD;            // row stride D3
    const bf16* Kp = qkv + (size_t)b * S * D3 + Dm + h * HD;
    const bf16* Vt = vt + ((size_t)(b * H + h)) * HD * S;          // [64][S]
    const float scale = 0.125f;                                    // 1/sqrt(64)

    v16bf qa0 = load_a_frag(Q, D3, q0, 0, lane);
    v16bf qa1 = load_a_frag(Q, D3, q0, 32, lane);

    float mr[8], lr[8];
    v8f o[4];
#pragma unroll
    for (int r = 0; r < 8; ++r) { mr[r] = -1.0e30f; lr[r] = 0.0f; }
#pragma unroll
    for (int t = 0; t < 4; ++t) o[t] = vzero8();

    int col = lane & 15, half = lane >> 4;

    for (int j0 = 0; j0 < q0 + 16; j0 += 32) {
        // scores: two 16x16 tiles (keys j0..j0+15, j0+16..j0+31)
        v8f s0 = vzero8(), s1 = vzero8();
        v16bf kb;
        kb = load_b_frag(Kp, D3, j0,      0,  lane, S); s0 = wmma_bf16(qa0, kb, s0);
        kb = load_b_frag(Kp, D3, j0,      32, lane, S); s0 = wmma_bf16(qa1, kb, s0);
        kb = load_b_frag(Kp, D3, j0 + 16, 0,  lane, S); s1 = wmma_bf16(qa0, kb, s1);
        kb = load_b_frag(Kp, D3, j0 + 16, 32, lane, S); s1 = wmma_bf16(qa1, kb, s1);

        // online softmax per row (rows live across the 16 lanes of each half-wave)
#pragma unroll
        for (int r = 0; r < 8; ++r) {
            int qrow = q0 + r + half * 8;
            float v0 = (j0 + col      <= qrow) ? s0[r] * scale : -1.0e30f;
            float v1 = (j0 + 16 + col <= qrow) ? s1[r] * scale : -1.0e30f;
            float rm = fmaxf(v0, v1);
#pragma unroll
            for (int m = 1; m < 16; m <<= 1) rm = fmaxf(rm, __shfl_xor(rm, m, 32));
            float mn = fmaxf(mr[r], rm);
            float p0 = __expf(v0 - mn);
            float p1 = __expf(v1 - mn);
            float rs = p0 + p1;
#pragma unroll
            for (int m = 1; m < 16; m <<= 1) rs += __shfl_xor(rs, m, 32);
            float sc = __expf(mr[r] - mn);
            lr[r] = lr[r] * sc + rs;
            mr[r] = mn;
#pragma unroll
            for (int t = 0; t < 4; ++t) o[t][r] *= sc;
            int prow = r + half * 8;
            Pst[wave][prow][col]      = (bf16)p0;
            Pst[wave][prow][16 + col] = (bf16)p1;
        }
        asm volatile("s_wait_dscnt 0x0" ::: "memory");  // LDS stores visible to cross-lane reads

        // P (16x32) as A-fragment from LDS; O += P * V (four 16-dim chunks)
        v16bf pa = load_a_frag(&Pst[wave][0][0], 32, 0, 0, lane);
#pragma unroll
        for (int t = 0; t < 4; ++t) {
            v16bf vb = load_b_frag(Vt, S, t * 16, j0, lane, HD);
            o[t] = wmma_bf16(pa, vb, o[t]);
        }
    }

    // normalize and write ctx[b, q, h*64 + d] (bf16, feeds out-proj GEMM)
#pragma unroll
    for (int t = 0; t < 4; ++t) {
#pragma unroll
        for (int r = 0; r < 8; ++r) {
            int m = q0 + r + half * 8;
            ctx[((size_t)(b * S + m)) * Dm + h * HD + t * 16 + col] = (bf16)(o[t][r] / lr[r]);
        }
    }
}

// ---------------------------------------------------------------------------
extern "C" void kernel_launch(void* const* d_in, const int* in_sizes, int n_in,
                              void* d_out, int out_size, void* d_ws, size_t ws_size,
                              hipStream_t stream) {
    const int B = 2, S = 2048, D = 768, L = 12, V = 50257;
    const int D3 = 3 * D, D4 = 4 * D, T = B * S, H = 12;

    const int*   idx  = (const int*)d_in[0];
    const float* wte  = (const float*)d_in[1];
    const float* wpe  = (const float*)d_in[2];
    const float* ln1w = (const float*)d_in[3];
    const float* ln1b = (const float*)d_in[4];
    const float* qkvw = (const float*)d_in[5];
    const float* outw = (const float*)d_in[6];
    const float* ln2w = (const float*)d_in[7];
    const float* ln2b = (const float*)d_in[8];
    const float* f1w  = (const float*)d_in[9];
    const float* f2w  = (const float*)d_in[10];
    const float* lnfw = (const float*)d_in[11];
    const float* lnfb = (const float*)d_in[12];
    float* out = (float*)d_out;

    size_t off = 0;
    auto alloc = [&](size_t bytes) -> void* {
        off = (off + 255) & ~(size_t)255;
        void* p = (char*)d_ws + off;
        off += bytes;
        return p;
    };

    bf16* wteb  = (bf16*)alloc((size_t)V * D * 2);
    bf16* qkvwb = (bf16*)alloc((size_t)L * D3 * D * 2);
    bf16* outwb = (bf16*)alloc((size_t)L * D * D * 2);
    bf16* f1b   = (bf16*)alloc((size_t)L * D4 * D * 2);
    bf16* f2b   = (bf16*)alloc((size_t)L * D * D4 * 2);
    float* x    = (float*)alloc((size_t)T * D * 4);
    bf16* hbuf  = (bf16*)alloc((size_t)T * D * 2);
    bf16* qkvb  = (bf16*)alloc((size_t)T * D3 * 2);
    bf16* vtb   = (bf16*)alloc((size_t)B * H * 64 * S * 2);
    bf16* ctxb  = (bf16*)alloc((size_t)T * D * 2);
    bf16* actb  = (bf16*)alloc((size_t)T * D4 * 2);

    auto cvt = [&](const float* src, bf16* dst, size_t n) {
        cvt_kernel<<<dim3((unsigned)((n + 255) / 256)), 256, 0, stream>>>(src, dst, n);
    };
    cvt(wte,  wteb,  (size_t)V * D);
    cvt(qkvw, qkvwb, (size_t)L * D3 * D);
    cvt(outw, outwb, (size_t)L * D * D);
    cvt(f1w,  f1b,   (size_t)L * D4 * D);
    cvt(f2w,  f2b,   (size_t)L * D * D4);

    embed_kernel<<<dim3(T * D / 256), 256, 0, stream>>>(idx, wte, wpe, x, S, D);

    for (int l = 0; l < L; ++l) {
        ln_kernel<<<dim3(T / 8), 256, 0, stream>>>(x, ln1w + l * D, ln1b + l * D, hbuf, D);
        gemm_bf16_kernel<<<dim3(D3 / 64, T / 256), 256, 0, stream>>>(
            hbuf, D, qkvwb + (size_t)l * D3 * D, D, D3, D, nullptr, qkvb, D3, 0);
        vtrans_kernel<<<dim3(B * H * 64 * S / 256), 256, 0, stream>>>(qkvb, vtb, S);
        attn_kernel<<<dim3(S / 64, H, B), 128, 0, stream>>>(qkvb, vtb, ctxb, S);
        gemm_bf16_kernel<<<dim3(D / 64, T / 256), 256, 0, stream>>>(
            ctxb, D, outwb + (size_t)l * D * D, D, D, D, x, nullptr, D, 2);
        ln_kernel<<<dim3(T / 8), 256, 0, stream>>>(x, ln2w + l * D, ln2b + l * D, hbuf, D);
        gemm_bf16_kernel<<<dim3(D4 / 64, T / 256), 256, 0, stream>>>(
            hbuf, D, f1b + (size_t)l * D4 * D, D, D4, D, nullptr, actb, D4, 3);
        gemm_bf16_kernel<<<dim3(D / 64, T / 256), 256, 0, stream>>>(
            actb, D4, f2b + (size_t)l * D * D4, D4, D, D4, x, nullptr, D, 2);
    }

    ln_kernel<<<dim3(T / 8), 256, 0, stream>>>(x, lnfw, lnfb, hbuf, D);
    gemm_bf16_kernel<<<dim3((V + 63) / 64, T / 256), 256, 0, stream>>>(
        hbuf, D, wteb, D, V, D, out, nullptr, V, 1);
}
